// GNNPolicy_51367808860366
// MI455X (gfx1250) — matrix operand order repaired and checked by reading, not compiled, not run on hardware
//
#include <hip/hip_runtime.h>

// ---------------------------------------------------------------------------
// GNN bipartite-conv forward for MI455X (gfx1250, wave32, WMMA bf16).
// All GEMMs use v_wmma_f32_16x16x32_bf16 with 16-row wave tiles.
// Edge pipeline (the bandwidth-dominant part) is fully fused:
//   edgesB @ We  (+gather ri/lj)  -> LN -> leaky -> @ Wf + bf -> atomic scatter
// ---------------------------------------------------------------------------

#define LEAKY_SLOPE 0.01f
#define EMBD 128
#define HIDD 256

typedef unsigned short u16;
typedef __attribute__((ext_vector_type(16))) __bf16 v16bf;
typedef __attribute__((ext_vector_type(8)))  float  v8f;

union FragU { v16bf v; uint4 q[2]; };

__device__ __forceinline__ u16 f2bf(float f) {            // RNE float -> bf16
  unsigned int u = __float_as_uint(f);
  u += 0x7fffu + ((u >> 16) & 1u);
  return (u16)(u >> 16);
}

__device__ __forceinline__ float leaky(float x) { return x > 0.f ? x : LEAKY_SLOPE * x; }
__device__ __forceinline__ int imin(int a, int b) { return a < b ? a : b; }

// A fragment (16x32 bf16): lane(half,m) holds K = half*16+{0..15}
__device__ __forceinline__ v16bf load_fragA(const u16* X, int ld, int m, int kbase, int half) {
  FragU u;
  const uint4* p = (const uint4*)(X + (size_t)m * ld + kbase + half * 16);
  u.q[0] = p[0];
  u.q[1] = p[1];   // +8 elements
  return u.v;
}

// B fragment (32x16 bf16) from transposed weights Wt[Nout][K]: lane(half,n) holds K = half*16+{0..15}
__device__ __forceinline__ v16bf load_fragB(const u16* Wt, int ld, int n, int kbase, int half) {
  FragU u;
  const uint4* p = (const uint4*)(Wt + (size_t)n * ld + kbase + half * 16);
  u.q[0] = p[0];
  u.q[1] = p[1];
  return u.v;
}

__device__ __forceinline__ v8f wmma_bf16(v16bf a, v16bf b, v8f c) {
  return __builtin_amdgcn_wmma_f32_16x16x32_bf16(false, a, false, b, (short)0, c, false, false);
}

// ---------------------------------------------------------------------------
// Weight prep: f32 [K][N] -> bf16 transposed [N][Kpad] (zero padded K)
// ---------------------------------------------------------------------------
__global__ void prep_weight_kernel(const float* __restrict__ W, int K, int Nout, int Kpad,
                                   u16* __restrict__ Wt) {
  int idx = blockIdx.x * 256 + threadIdx.x;
  if (idx >= Nout * Kpad) return;
  int nf = idx / Kpad, k = idx % Kpad;
  Wt[idx] = (k < K) ? f2bf(W[(size_t)k * Nout + nf]) : (u16)0;
}

__global__ void zero_kernel(float* __restrict__ p, int n) {
  int i = blockIdx.x * 256 + threadIdx.x;
  if (i < n) p[i] = 0.f;
}

// ---------------------------------------------------------------------------
// Embedding MLP: LN(Fin) -> W0[Fin,256]+b0 -> leaky -> W1[256,128]+b1 -> leaky
// One wave per 16 rows, 4 waves / block.
// ---------------------------------------------------------------------------
__global__ __launch_bounds__(128) void emb_mlp_kernel(
    const float* __restrict__ X, int N, int Fin,
    const float* __restrict__ ln_g, const float* __restrict__ ln_b,
    const u16* __restrict__ W0t, const float* __restrict__ b0,
    const u16* __restrict__ W1t, const float* __restrict__ b1,
    float* __restrict__ outF, u16* __restrict__ outB) {
  __shared__ u16 lnT[4][16][32];     // LN output, K padded to 32
  __shared__ u16 hidT[4][16][HIDD];  // hidden (bf16)
  const int wave = threadIdx.x >> 5, lane = threadIdx.x & 31;
  const int half = lane >> 4, mn = lane & 15;
  const int rowBase = blockIdx.x * 64 + wave * 16;

  if (lane < 16) {
    int r = rowBase + lane;
    if (r < N) {
      float xs[32];
      float s = 0.f;
      for (int k = 0; k < Fin; ++k) { xs[k] = X[(size_t)r * Fin + k]; s += xs[k]; }
      float mean = s / (float)Fin;
      float v = 0.f;
      for (int k = 0; k < Fin; ++k) { float d = xs[k] - mean; v += d * d; }
      float rstd = rsqrtf(v / (float)Fin + 1e-5f);
      for (int k = 0; k < 32; ++k) {
        float y = (k < Fin) ? ((xs[k] - mean) * rstd * ln_g[k] + ln_b[k]) : 0.f;
        lnT[wave][lane][k] = f2bf(y);
      }
    } else {
      for (int k = 0; k < 32; ++k) lnT[wave][lane][k] = 0;
    }
  }
  __syncthreads();

  // GEMM1: [16x32] @ [32x256] (K = Fin padded to 32)
  for (int nt = 0; nt < HIDD / 16; ++nt) {
    float bz = b0[nt * 16 + mn];
    v8f acc;
#pragma unroll
    for (int i = 0; i < 8; ++i) acc[i] = bz;
    v16bf a = load_fragA(&lnT[wave][0][0], 32, mn, 0, half);
    v16bf b = load_fragB(W0t, 32, nt * 16 + mn, 0, half);
    acc = wmma_bf16(a, b, acc);
#pragma unroll
    for (int i = 0; i < 8; ++i)
      hidT[wave][i + 8 * half][nt * 16 + mn] = f2bf(leaky(acc[i]));
  }
  __syncthreads();

  // GEMM2: [16x256] @ [256x128]
  for (int nt = 0; nt < EMBD / 16; ++nt) {
    float bz = b1[nt * 16 + mn];
    v8f acc;
#pragma unroll
    for (int i = 0; i < 8; ++i) acc[i] = bz;
#pragma unroll
    for (int kc = 0; kc < HIDD / 32; ++kc) {
      v16bf a = load_fragA(&hidT[wave][0][0], HIDD, mn, kc * 32, half);
      v16bf b = load_fragB(W1t, HIDD, nt * 16 + mn, kc * 32, half);
      acc = wmma_bf16(a, b, acc);
    }
    int col = nt * 16 + mn;
#pragma unroll
    for (int i = 0; i < 8; ++i) {
      int row = rowBase + i + 8 * half;
      if (row < N) {
        float y = leaky(acc[i]);
        if (outF) outF[(size_t)row * EMBD + col] = y;
        if (outB) outB[(size_t)row * EMBD + col] = f2bf(y);
      }
    }
  }
}

// ---------------------------------------------------------------------------
// Node linear: Y[N,128] (f32) = Xb[N,128](bf16) @ W[128,128] (+bias)
// ---------------------------------------------------------------------------
__global__ __launch_bounds__(128) void node_linear_kernel(
    const u16* __restrict__ Xb, int N, const u16* __restrict__ Wt,
    const float* __restrict__ bias, float* __restrict__ Y) {
  const int wave = threadIdx.x >> 5, lane = threadIdx.x & 31;
  const int half = lane >> 4, mn = lane & 15;
  const int rowBase = blockIdx.x * 64 + wave * 16;
  const int arow = imin(rowBase + mn, N - 1);

  for (int nt = 0; nt < EMBD / 16; ++nt) {
    float bz = bias ? bias[nt * 16 + mn] : 0.f;
    v8f acc;
#pragma unroll
    for (int i = 0; i < 8; ++i) acc[i] = bz;
#pragma unroll
    for (int kc = 0; kc < EMBD / 32; ++kc) {
      v16bf a = load_fragA(Xb, EMBD, arow, kc * 32, half);
      v16bf b = load_fragB(Wt, EMBD, nt * 16 + mn, kc * 32, half);
      acc = wmma_bf16(a, b, acc);
    }
    int col = nt * 16 + mn;
#pragma unroll
    for (int i = 0; i < 8; ++i) {
      int row = rowBase + i + 8 * half;
      if (row < N) Y[(size_t)row * EMBD + col] = acc[i];
    }
  }
}

// ---------------------------------------------------------------------------
// Fused edge pipeline (per 16-edge wave tile):
//   z = edgesB @ We + ri[e_dst] + lj[e_src]   (WMMA, K=128)
//   m = leaky(LN(z))                          (LDS, shfl reduce)
//   msg = m @ Wf + bf                         (WMMA, K=128)
//   agg[e_dst] += msg                         (global_atomic_add_f32)
// ---------------------------------------------------------------------------
__global__ __launch_bounds__(128) void edge_fused_kernel(
    const u16* __restrict__ edgesB, const int* __restrict__ eSrc, const int* __restrict__ eDst,
    const float* __restrict__ lj, const float* __restrict__ ri,
    const u16* __restrict__ Wet, const u16* __restrict__ Wft,
    const float* __restrict__ bf, const float* __restrict__ ln1g, const float* __restrict__ ln1b,
    float* __restrict__ agg, int NE) {
  __shared__ float zt[4][16][EMBD];   // 32 KB
  __shared__ u16   mt[4][16][EMBD];   // 16 KB
  __shared__ int   sIdx[4][16], dIdx[4][16];
  const int wave = threadIdx.x >> 5, lane = threadIdx.x & 31;
  const int half = lane >> 4, mn = lane & 15;
  const int eBase = blockIdx.x * 64 + wave * 16;

  if (lane < 16) {
    int e = imin(eBase + lane, NE - 1);
    sIdx[wave][lane] = eSrc[e];
    dIdx[wave][lane] = eDst[e];
  }
  __syncthreads();

  int sArr[8], dArr[8];
#pragma unroll
  for (int i = 0; i < 8; ++i) {
    sArr[i] = sIdx[wave][i + 8 * half];
    dArr[i] = dIdx[wave][i + 8 * half];
  }
  const int aRow = imin(eBase + mn, NE - 1);

  // z = edgesB @ We + gathers
  for (int nt = 0; nt < EMBD / 16; ++nt) {
    v8f acc;
#pragma unroll
    for (int i = 0; i < 8; ++i) acc[i] = 0.f;
#pragma unroll
    for (int kc = 0; kc < EMBD / 32; ++kc) {
      v16bf a = load_fragA(edgesB, EMBD, aRow, kc * 32, half);
      v16bf b = load_fragB(Wet, EMBD, nt * 16 + mn, kc * 32, half);
      acc = wmma_bf16(a, b, acc);
    }
    int col = nt * 16 + mn;
#pragma unroll
    for (int i = 0; i < 8; ++i) {
      float z = acc[i] + lj[(size_t)sArr[i] * EMBD + col] + ri[(size_t)dArr[i] * EMBD + col];
      zt[wave][i + 8 * half][col] = z;
    }
  }
  __syncthreads();

  // LN + leaky -> bf16 (lane pair (m, m+16) cooperates on one row)
  {
    const float* row = &zt[wave][mn][0];
    int c0 = half * 64;
    float s = 0.f;
    for (int c = c0; c < c0 + 64; ++c) s += row[c];
    s += __shfl_xor(s, 16);
    float mean = s * (1.f / 128.f);
    float v = 0.f;
    for (int c = c0; c < c0 + 64; ++c) { float d = row[c] - mean; v += d * d; }
    v += __shfl_xor(v, 16);
    float rstd = rsqrtf(v * (1.f / 128.f) + 1e-5f);
    for (int c = c0; c < c0 + 64; ++c) {
      float y = (row[c] - mean) * rstd * ln1g[c] + ln1b[c];
      mt[wave][mn][c] = f2bf(leaky(y));
    }
  }
  __syncthreads();

  // msg = m @ Wf + bf ; scatter-add into agg
  for (int nt = 0; nt < EMBD / 16; ++nt) {
    int col = nt * 16 + mn;
    float bz = bf[col];
    v8f acc;
#pragma unroll
    for (int i = 0; i < 8; ++i) acc[i] = bz;
#pragma unroll
    for (int kc = 0; kc < EMBD / 32; ++kc) {
      v16bf a = load_fragA(&mt[wave][0][0], EMBD, mn, kc * 32, half);
      v16bf b = load_fragB(Wft, EMBD, col, kc * 32, half);
      acc = wmma_bf16(a, b, acc);
    }
#pragma unroll
    for (int i = 0; i < 8; ++i) {
      int e = eBase + i + 8 * half;
      if (e < NE)
        __hip_atomic_fetch_add(&agg[(size_t)dArr[i] * EMBD + col], acc[i],
                               __ATOMIC_RELAXED, __HIP_MEMORY_SCOPE_AGENT);
    }
  }
}

// ---------------------------------------------------------------------------
// Post-aggregation: h = [LN(agg), right] (256) -> @Wo1+bo1 -> leaky -> @Wo2+bo2 -> leaky
// ---------------------------------------------------------------------------
__global__ __launch_bounds__(128) void post_agg_kernel(
    const float* __restrict__ agg, const float* __restrict__ right,
    const float* __restrict__ ln2g, const float* __restrict__ ln2b,
    const u16* __restrict__ Wo1t, const float* __restrict__ bo1,
    const u16* __restrict__ Wo2t, const float* __restrict__ bo2,
    float* __restrict__ outF, u16* __restrict__ outB, int N) {
  __shared__ u16 ht[4][16][2 * EMBD];  // 32 KB
  __shared__ u16 h2[4][16][EMBD];      // 16 KB
  const int wave = threadIdx.x >> 5, lane = threadIdx.x & 31;
  const int half = lane >> 4, mn = lane & 15;
  const int rowBase = blockIdx.x * 64 + wave * 16;

  {
    int row = imin(rowBase + mn, N - 1);
    const float* arow = agg + (size_t)row * EMBD;
    int c0 = half * 64;
    float s = 0.f;
    for (int c = c0; c < c0 + 64; ++c) s += arow[c];
    s += __shfl_xor(s, 16);
    float mean = s * (1.f / 128.f);
    float v = 0.f;
    for (int c = c0; c < c0 + 64; ++c) { float d = arow[c] - mean; v += d * d; }
    v += __shfl_xor(v, 16);
    float rstd = rsqrtf(v * (1.f / 128.f) + 1e-5f);
    for (int c = c0; c < c0 + 64; ++c)
      ht[wave][mn][c] = f2bf((arow[c] - mean) * rstd * ln2g[c] + ln2b[c]);
    const float* rrow = right + (size_t)row * EMBD;
    for (int c = c0; c < c0 + 64; ++c)
      ht[wave][mn][EMBD + c] = f2bf(rrow[c]);
  }
  __syncthreads();

  // GEMM1: [16x256] @ [256x128]
  for (int nt = 0; nt < EMBD / 16; ++nt) {
    float bz = bo1[nt * 16 + mn];
    v8f acc;
#pragma unroll
    for (int i = 0; i < 8; ++i) acc[i] = bz;
#pragma unroll
    for (int kc = 0; kc < (2 * EMBD) / 32; ++kc) {
      v16bf a = load_fragA(&ht[wave][0][0], 2 * EMBD, mn, kc * 32, half);
      v16bf b = load_fragB(Wo1t, 2 * EMBD, nt * 16 + mn, kc * 32, half);
      acc = wmma_bf16(a, b, acc);
    }
#pragma unroll
    for (int i = 0; i < 8; ++i)
      h2[wave][i + 8 * half][nt * 16 + mn] = f2bf(leaky(acc[i]));
  }
  __syncthreads();

  // GEMM2: [16x128] @ [128x128]
  for (int nt = 0; nt < EMBD / 16; ++nt) {
    float bz = bo2[nt * 16 + mn];
    v8f acc;
#pragma unroll
    for (int i = 0; i < 8; ++i) acc[i] = bz;
#pragma unroll
    for (int kc = 0; kc < EMBD / 32; ++kc) {
      v16bf a = load_fragA(&h2[wave][0][0], EMBD, mn, kc * 32, half);
      v16bf b = load_fragB(Wo2t, EMBD, nt * 16 + mn, kc * 32, half);
      acc = wmma_bf16(a, b, acc);
    }
    int col = nt * 16 + mn;
#pragma unroll
    for (int i = 0; i < 8; ++i) {
      int row = rowBase + i + 8 * half;
      if (row < N) {
        float y = leaky(acc[i]);
        if (outF) outF[(size_t)row * EMBD + col] = y;
        if (outB) outB[(size_t)row * EMBD + col] = f2bf(y);
      }
    }
  }
}

// ---------------------------------------------------------------------------
// Host orchestration
// ---------------------------------------------------------------------------
struct EmbW { const float *ln_g, *ln_b, *b0, *b1; u16 *W0t, *W1t; int Fin; };
struct ConvW {
  u16 *Wet, *Wft, *Wlt, *Wrt, *Wo1t, *Wo2t;
  const float *bf, *bl, *bo1, *bo2, *ln1g, *ln1b, *ln2g, *ln2b;
};

extern "C" void kernel_launch(void* const* d_in, const int* in_sizes, int n_in,
                              void* d_out, int out_size, void* d_ws, size_t ws_size,
                              hipStream_t stream) {
  (void)in_sizes; (void)out_size; (void)ws_size;
  if (n_in < 79) return;  // expected 5 tensors + 74 param leaves

  const int NITEMS = 40000, NBOXES = 8000, NEDGES = 400000;
  const float* itemX = (const float*)d_in[0];
  const float* boxX  = (const float*)d_in[1];
  const float* edgeX = (const float*)d_in[2];
  const int*   eSrc  = (const int*)d_in[3];
  const int*   eDst  = (const int*)d_in[4];

  // Param flattening assumption: jax pytree order (dict keys sorted):
  //   params: box_emb@5, edge_emb@11, item_emb@17,
  //           layers[0]{b2i@23, i2b@37}, layers[1]{b2i@51, i2b@65}
  //   emb:  W0, W1, b0, b1, ln_b, ln_g
  //   conv: We, Wf, Wl, Wo1, Wo2, Wr, bf, bl, bo1, bo2, ln1_b, ln1_g, ln2_b, ln2_g
  auto F = [&](int i) { return (const float*)d_in[i]; };

  size_t cur = 0;
  auto alloc = [&](size_t bytes) -> void* {
    cur = (cur + 255) & ~(size_t)255;
    void* p = (char*)d_ws + cur;
    cur += bytes;
    return p;
  };

  float* itemsF = (float*)alloc((size_t)NITEMS * EMBD * 4);
  float* boxesF = (float*)alloc((size_t)NBOXES * EMBD * 4);
  u16*   itemsB = (u16*)  alloc((size_t)NITEMS * EMBD * 2);
  u16*   boxesB = (u16*)  alloc((size_t)NBOXES * EMBD * 2);
  u16*   edgesB = (u16*)  alloc((size_t)NEDGES * EMBD * 2);
  float* riBuf  = (float*)alloc((size_t)NITEMS * EMBD * 4);
  float* ljBuf  = (float*)alloc((size_t)NITEMS * EMBD * 4);
  float* aggBuf = (float*)alloc((size_t)NITEMS * EMBD * 4);

  auto prep = [&](const float* W, int K, int Nout, int Kpad, u16* Wt) {
    int total = Nout * Kpad;
    prep_weight_kernel<<<(total + 255) / 256, 256, 0, stream>>>(W, K, Nout, Kpad, Wt);
  };

  auto setup_emb = [&](int base, int Fin) {
    EmbW e; e.Fin = Fin;
    e.W0t = (u16*)alloc((size_t)HIDD * 32 * 2);
    e.W1t = (u16*)alloc((size_t)EMBD * HIDD * 2);
    prep(F(base + 0), Fin, HIDD, 32, e.W0t);
    prep(F(base + 1), HIDD, EMBD, HIDD, e.W1t);
    e.b0 = F(base + 2); e.b1 = F(base + 3);
    e.ln_b = F(base + 4); e.ln_g = F(base + 5);
    return e;
  };
  auto setup_conv = [&](int b) {
    ConvW c;
    c.Wet  = (u16*)alloc((size_t)EMBD * EMBD * 2);
    c.Wft  = (u16*)alloc((size_t)EMBD * EMBD * 2);
    c.Wlt  = (u16*)alloc((size_t)EMBD * EMBD * 2);
    c.Wo1t = (u16*)alloc((size_t)EMBD * 2 * EMBD * 2);
    c.Wo2t = (u16*)alloc((size_t)EMBD * EMBD * 2);
    c.Wrt  = (u16*)alloc((size_t)EMBD * EMBD * 2);
    prep(F(b + 0), EMBD, EMBD, EMBD, c.Wet);
    prep(F(b + 1), EMBD, EMBD, EMBD, c.Wft);
    prep(F(b + 2), EMBD, EMBD, EMBD, c.Wlt);
    prep(F(b + 3), 2 * EMBD, EMBD, 2 * EMBD, c.Wo1t);
    prep(F(b + 4), EMBD, EMBD, EMBD, c.Wo2t);
    prep(F(b + 5), EMBD, EMBD, EMBD, c.Wrt);
    c.bf = F(b + 6); c.bl = F(b + 7); c.bo1 = F(b + 8); c.bo2 = F(b + 9);
    c.ln1b = F(b + 10); c.ln1g = F(b + 11); c.ln2b = F(b + 12); c.ln2g = F(b + 13);
    return c;
  };

  EmbW boxE  = setup_emb(5, 16);
  EmbW edgeE = setup_emb(11, 8);
  EmbW itemE = setup_emb(17, 32);
  ConvW c0b2i = setup_conv(23), c0i2b = setup_conv(37);
  ConvW c1b2i = setup_conv(51), c1i2b = setup_conv(65);

  // Embedding MLPs
  emb_mlp_kernel<<<(NITEMS + 63) / 64, 128, 0, stream>>>(
      itemX, NITEMS, itemE.Fin, itemE.ln_g, itemE.ln_b,
      itemE.W0t, itemE.b0, itemE.W1t, itemE.b1, itemsF, itemsB);
  emb_mlp_kernel<<<(NBOXES + 63) / 64, 128, 0, stream>>>(
      boxX, NBOXES, boxE.Fin, boxE.ln_g, boxE.ln_b,
      boxE.W0t, boxE.b0, boxE.W1t, boxE.b1, boxesF, boxesB);
  emb_mlp_kernel<<<(NEDGES + 63) / 64, 128, 0, stream>>>(
      edgeX, NEDGES, edgeE.Fin, edgeE.ln_g, edgeE.ln_b,
      edgeE.W0t, edgeE.b0, edgeE.W1t, edgeE.b1, (float*)nullptr, edgesB);

  auto run_conv = [&](const u16* leftB, const u16* rightB, const float* rightF,
                      const int* es, const int* ed, int nLeft, int nRight,
                      const ConvW& w, float* outF, u16* outB) {
    node_linear_kernel<<<(nRight + 63) / 64, 128, 0, stream>>>(rightB, nRight, w.Wlt, w.bl, riBuf);
    node_linear_kernel<<<(nLeft + 63) / 64, 128, 0, stream>>>(leftB, nLeft, w.Wrt, nullptr, ljBuf);
    zero_kernel<<<(nRight * EMBD + 255) / 256, 256, 0, stream>>>(aggBuf, nRight * EMBD);
    edge_fused_kernel<<<(NEDGES + 63) / 64, 128, 0, stream>>>(
        edgesB, es, ed, ljBuf, riBuf, w.Wet, w.Wft, w.bf, w.ln1g, w.ln1b, aggBuf, NEDGES);
    post_agg_kernel<<<(nRight + 63) / 64, 128, 0, stream>>>(
        aggBuf, rightF, w.ln2g, w.ln2b, w.Wo1t, w.bo1, w.Wo2t, w.bo2, outF, outB, nRight);
  };

  // layer 0: items -> boxes (i2b), then boxes_new -> items (b2i, reversed edges)
  run_conv(itemsB, boxesB, boxesF, eSrc, eDst, NITEMS, NBOXES, c0i2b, boxesF, boxesB);
  run_conv(boxesB, itemsB, itemsF, eDst, eSrc, NBOXES, NITEMS, c0b2i, itemsF, itemsB);
  // layer 1
  run_conv(itemsB, boxesB, boxesF, eSrc, eDst, NITEMS, NBOXES, c1i2b, boxesF, boxesB);
  run_conv(boxesB, itemsB, itemsF, eDst, eSrc, NBOXES, NITEMS, c1b2i, itemsF, itemsB);

  // outputs: (items, boxes) concatenated
  hipMemcpyAsync(d_out, itemsF, (size_t)NITEMS * EMBD * 4, hipMemcpyDeviceToDevice, stream);
  hipMemcpyAsync((float*)d_out + (size_t)NITEMS * EMBD, boxesF, (size_t)NBOXES * EMBD * 4,
                 hipMemcpyDeviceToDevice, stream);
}
